// LSTM_80882824118984
// MI455X (gfx1250) — matrix operand order repaired and checked by reading, not compiled, not run on hardware
//
#include <hip/hip_runtime.h>

// ---------------------------------------------------------------------------
// LSTM on MI455X (gfx1250, wave32, WMMA).
//
//  * Per timestep: z = [x_t ; h_{t-1}] @ [Wx;Wh]  (K=1280, N=4096), bf16
//    operands, f32 accumulate via v_wmma_f32_16x16x32_bf16.
//  * Packed weights (10.5 MB bf16) stay resident in the 192 MB L2 across all
//    512 steps; each step streams them at L2 bandwidth.
//  * Each wave owns one 16x16 (batch,hidden) tile for ALL FOUR gates, so the
//    gate nonlinearity + c/h update are wave-local.
//  * K-loop fully unrolled; software pipeline ENFORCED with
//    __builtin_amdgcn_sched_group_barrier (0x20=VMEM read, 0x8=WMMA):
//    20 loads up front, then 38 x {4 WMMA ; 10 loads}, then 8 WMMA.
//    This keeps 2 fragment stages in flight so WMMAs wait on loadcnt<=N
//    instead of draining to 0 (only ~1 wave/SIMD exists to hide latency).
// ---------------------------------------------------------------------------

typedef __attribute__((ext_vector_type(16))) __bf16 v16bf;
typedef __attribute__((ext_vector_type(8)))  __bf16 v8bf;
typedef __attribute__((ext_vector_type(8)))  float  v8f;

#define BSZ   64
#define TLEN  512
#define IIN   256
#define HD    1024
#define OD    10
#define KTOT  1280            // I + H
#define NKT   40              // KTOT / 32
#define GSTR  ((size_t)64 * NKT * 512)  // per-gate stride in packed W (halves)

// ---- init: zero c state and h0 buffer (ws is poisoned once by harness) ----
__global__ void zero_state(float* __restrict__ c, __bf16* __restrict__ h0) {
    int i = blockIdx.x * blockDim.x + threadIdx.x;
    if (i < BSZ * HD) { c[i] = 0.0f; h0[i] = (__bf16)0.0f; }
}

// ---- x [B][T][I] f32  ->  xbf [T][B][I] bf16 (A-matrix row-major per t) ---
__global__ void conv_x(const float* __restrict__ x, __bf16* __restrict__ xbf) {
    int id = blockIdx.x * blockDim.x + threadIdx.x;   // ((t*64+b)*256+i)
    int i  = id & 255;
    int b  = (id >> 8) & 63;
    int t  = id >> 14;
    xbf[id] = (__bf16)x[((size_t)b * TLEN + t) * IIN + i];
}

// ---- pack Wx(4,256,1024) + Wh(4,1024,1024) f32 -> bf16 B-fragment order ---
// Layout: wpk[ntile(=gate*64+nt)][kt(0..39)][lane(0..31)][j(0..15)]
// B 32x16 bf16 fragment: element j of lane  ->  N = nt*16 + (lane&15),
//                                              K = kt*32 + (lane&16 ? 8:0) + (j&7) + (j&8 ? 16:0)
__global__ void pack_w(const float* __restrict__ Wx, const float* __restrict__ Wh,
                       __bf16* __restrict__ wpk) {
    int id   = blockIdx.x * blockDim.x + threadIdx.x;
    int j    = id & 15;
    int lane = (id >> 4) & 31;
    int kt   = (id >> 9) % NKT;
    int ntile = id / (512 * NKT);
    if (ntile >= 256) return;
    int k    = ntile >> 6;                       // gate
    int ncol = ((ntile & 63) << 4) + (lane & 15);
    int K    = kt * 32 + ((lane & 16) ? 8 : 0) + (j & 7) + ((j & 8) ? 16 : 0);
    float v  = (K < IIN) ? Wx[((size_t)k * IIN + K) * HD + ncol]
                         : Wh[((size_t)k * HD + (K - IIN)) * HD + ncol];
    wpk[id] = (__bf16)v;
}

__device__ __forceinline__ float sigm(float x) {
    return 1.0f / (1.0f + __expf(-x));
}

// One fragment stage: A (two 8-half runs) + B for all 4 gates.
struct Frag {
    v8bf  alo, ahi;
    v16bf b0, b1, b2, b3;
};

__device__ __forceinline__ Frag ldfrag(const __bf16* ap, const __bf16* bp) {
    Frag f;
    f.alo = *(const v8bf*)(ap);
    f.ahi = *(const v8bf*)(ap + 16);
    f.b0  = *(const v16bf*)(bp);
    f.b1  = *(const v16bf*)(bp + GSTR);
    f.b2  = *(const v16bf*)(bp + 2 * GSTR);
    f.b3  = *(const v16bf*)(bp + 3 * GSTR);
    return f;
}

__device__ __forceinline__ void domac(const Frag& f, v8f& a0, v8f& a1,
                                      v8f& a2, v8f& a3) {
    v16bf a;
#pragma unroll
    for (int j = 0; j < 8; ++j) { a[j] = f.alo[j]; a[j + 8] = f.ahi[j]; }
    a0 = __builtin_amdgcn_wmma_f32_16x16x32_bf16(false, a, false, f.b0,
                                                 (short)0, a0, false, false);
    a1 = __builtin_amdgcn_wmma_f32_16x16x32_bf16(false, a, false, f.b1,
                                                 (short)0, a1, false, false);
    a2 = __builtin_amdgcn_wmma_f32_16x16x32_bf16(false, a, false, f.b2,
                                                 (short)0, a2, false, false);
    a3 = __builtin_amdgcn_wmma_f32_16x16x32_bf16(false, a, false, f.b3,
                                                 (short)0, a3, false, false);
}

// ---- one LSTM timestep: 256 waves, each owns a 16x16 (batch,hidden) tile --
__global__ void __launch_bounds__(128, 1)
lstm_step(const __bf16* __restrict__ xt,    // [64][256]  bf16 (this timestep)
          const __bf16* __restrict__ wpk,   // packed Wcat fragments
          const __bf16* __restrict__ hin,   // [64][1024] bf16 (h_{t-1})
          __bf16*       __restrict__ hout,  // [64][1024] bf16 (h_t)
          float*        __restrict__ cst,   // [64][1024] f32  (c state)
          const float*  __restrict__ bx,    // [4][1024]
          const float*  __restrict__ bh)    // [4][1024]
{
    const int lane = threadIdx.x & 31;
    const int wid  = (blockIdx.x * blockDim.x + threadIdx.x) >> 5;   // 0..255
    const int m0   = (wid >> 6) << 4;     // batch tile base  (0,16,32,48)
    const int ntl  = wid & 63;            // hidden tile index
    const int n0   = ntl << 4;

    // A-fragment addressing (16x32 bf16): lane holds row M = m0 + (lane&15),
    // two contiguous 8-half runs at K = base and K = base+16,
    // base = kt*32 + (lane>=16 ? 8 : 0).
    const int M   = m0 + (lane & 15);
    const int hiK = (lane & 16) ? 8 : 0;

    const __bf16* xrow = xt  + (size_t)M * IIN + hiK;   // kt = 0..7
    const __bf16* hrow = hin + (size_t)M * HD  + hiK;   // kt = 8..39
    const __bf16* bp0  = wpk + (size_t)ntl * NKT * 512 + (size_t)lane * 16;

    // kt is compile-time below (full unroll) -> these selects fold away.
    auto aPtr = [&](int kt) -> const __bf16* {
        return (kt < 8) ? (xrow + kt * 32) : (hrow + (kt - 8) * 32);
    };
    auto bPtr = [&](int kt) -> const __bf16* {
        return bp0 + (size_t)kt * 512;
    };

    v8f acc0 = {}, acc1 = {}, acc2 = {}, acc3 = {};

    // ---- fully unrolled, distance-2 rotating stage buffer ----
    Frag s0 = ldfrag(aPtr(0), bPtr(0));
    Frag s1 = ldfrag(aPtr(1), bPtr(1));

#pragma unroll
    for (int kt = 0; kt < NKT; kt += 2) {
        domac(s0, acc0, acc1, acc2, acc3);
        if (kt + 2 < NKT) s0 = ldfrag(aPtr(kt + 2), bPtr(kt + 2));
        domac(s1, acc0, acc1, acc2, acc3);
        if (kt + 3 < NKT) s1 = ldfrag(aPtr(kt + 3), bPtr(kt + 3));
    }

    // ---- enforce the software pipeline at the instruction level ----
    // Groups are matched in order within this scheduling region:
    //   20 VMEM-reads (kt=0,1 prologue), then for kt=0..37:
    //   {4 WMMA (kt) ; 10 VMEM-reads (kt+2)}, then 8 WMMA (kt=38,39).
    __builtin_amdgcn_sched_group_barrier(0x020, 20, 0);
#pragma unroll
    for (int i = 0; i < 38; ++i) {
        __builtin_amdgcn_sched_group_barrier(0x008, 4, 0);   // WMMA x4
        __builtin_amdgcn_sched_group_barrier(0x020, 10, 0);  // loads x10
    }
    __builtin_amdgcn_sched_group_barrier(0x008, 8, 0);       // drain WMMAs

    // C/D tile layout: VGPR r of lane -> row M = m0 + r + (lane>=16 ? 8 : 0),
    //                                    col N = n0 + (lane&15).
    const int n = n0 + (lane & 15);
    const float biasg = bx[0 * HD + n] + bh[0 * HD + n];
    const float biasi = bx[1 * HD + n] + bh[1 * HD + n];
    const float biasf = bx[2 * HD + n] + bh[2 * HD + n];
    const float biaso = bx[3 * HD + n] + bh[3 * HD + n];
    const int rbase = m0 + ((lane & 16) ? 8 : 0);

#pragma unroll
    for (int r = 0; r < 8; ++r) {
        const float g  = tanhf(acc0[r] + biasg);
        const float ig = sigm(acc1[r] + biasi);
        const float fg = sigm(acc2[r] + biasf);
        const float og = sigm(acc3[r] + biaso);
        const size_t idx = (size_t)(rbase + r) * HD + n;
        const float cn = g * ig + cst[idx] * fg;
        cst[idx]  = cn;
        hout[idx] = (__bf16)(tanhf(cn) * og);
    }
}

// ---- final projection (64x1024 @ 1024x10) + softmax over O=10 -------------
__global__ void proj_softmax(const __bf16* __restrict__ h,
                             const float* __restrict__ Wp,
                             const float* __restrict__ bp,
                             float* __restrict__ out) {
    __shared__ float p[OD];
    const int b = blockIdx.x;
    const int t = threadIdx.x;
    if (t < OD) {
        float acc = bp[t];
        for (int k = 0; k < HD; ++k)
            acc += (float)h[(size_t)b * HD + k] * Wp[(size_t)k * OD + t];
        p[t] = acc;
    }
    __syncthreads();
    if (t < OD) {
        float m = p[0];
#pragma unroll
        for (int j = 1; j < OD; ++j) m = fmaxf(m, p[j]);
        float s = 0.0f;
#pragma unroll
        for (int j = 0; j < OD; ++j) s += __expf(p[j] - m);
        out[b * OD + t] = __expf(p[t] - m) / s;
    }
}

extern "C" void kernel_launch(void* const* d_in, const int* in_sizes, int n_in,
                              void* d_out, int out_size, void* d_ws, size_t ws_size,
                              hipStream_t stream) {
    (void)in_sizes; (void)n_in; (void)out_size; (void)ws_size;

    const float* x  = (const float*)d_in[0];   // [64][512][256]
    const float* Wx = (const float*)d_in[1];   // [4][256][1024]
    const float* bx = (const float*)d_in[2];   // [4][1024]
    const float* Wh = (const float*)d_in[3];   // [4][1024][1024]
    const float* bh = (const float*)d_in[4];   // [4][1024]
    const float* Wp = (const float*)d_in[5];   // [1024][10]
    const float* bp = (const float*)d_in[6];   // [10]
    float* out = (float*)d_out;                // [64][10]

    // Workspace carve-out (~27.8 MB total)
    char* ws = (char*)d_ws;
    size_t off = 0;
    float*  cst = (float*)(ws + off);  off += (size_t)BSZ * HD * sizeof(float);
    __bf16* h0  = (__bf16*)(ws + off); off += (size_t)BSZ * HD * sizeof(__bf16);
    __bf16* h1  = (__bf16*)(ws + off); off += (size_t)BSZ * HD * sizeof(__bf16);
    __bf16* xbf = (__bf16*)(ws + off); off += (size_t)TLEN * BSZ * IIN * sizeof(__bf16);
    __bf16* wpk = (__bf16*)(ws + off);

    // State must be re-zeroed every call (harness does not re-poison).
    zero_state<<<(BSZ * HD + 255) / 256, 256, 0, stream>>>(cst, h0);

    // x -> bf16, [T][B][I] layout.
    conv_x<<<(TLEN * BSZ * IIN) / 256, 256, 0, stream>>>(x, xbf);

    // Pack [Wx;Wh] into WMMA B-fragment order.
    pack_w<<<(256 * NKT * 32 * 16) / 256, 256, 0, stream>>>(Wx, Wh, wpk);

    // Sequential recurrence: 512 dependent step kernels, stream-ordered.
    for (int t = 0; t < TLEN; ++t) {
        const __bf16* hi = (t & 1) ? h1 : h0;
        __bf16*       ho = (t & 1) ? h0 : h1;
        lstm_step<<<64, 128, 0, stream>>>(xbf + (size_t)t * BSZ * IIN,
                                          wpk, hi, ho, cst, bx, bh);
    }
    // t=511 (odd) wrote h0 -> final hidden state lives in h0.
    proj_softmax<<<BSZ, 32, 0, stream>>>(h0, Wp, bp, out);
}
